// SparseGatingNetwork_2250562863552
// MI455X (gfx1250) — matrix operand order repaired and checked by reading, not compiled, not run on hardware
//
#include <hip/hip_runtime.h>
#include <hip/hip_bf16.h>

#define B_ROWS 65536
#define D_IN   2048
#define H_MID  1024
#define E_EXP  64
#define K_TOP  8
#define LN_EPSF 1e-5f

typedef __attribute__((ext_vector_type(16))) __bf16 v16bf;
typedef __attribute__((ext_vector_type(8)))  float  v8f;
typedef __attribute__((ext_vector_type(4)))  float  v4f;
typedef __attribute__((ext_vector_type(4)))  int    v4i;

#define AS1 __attribute__((address_space(1)))
#define AS3 __attribute__((address_space(3)))

// CDNA5 async global->LDS path (ASYNCcnt); guarded so compile never regresses.
#if defined(__AMDGCN__) && __has_builtin(__builtin_amdgcn_global_load_async_to_lds_b128) && \
    __has_builtin(__builtin_amdgcn_s_wait_asynccnt)
#define USE_ASYNC_LDS 1
#else
#define USE_ASYNC_LDS 0
#endif

#if defined(__AMDGCN__) && __has_builtin(__builtin_amdgcn_sched_barrier)
#define SCHED_FENCE() __builtin_amdgcn_sched_barrier(0)
#else
#define SCHED_FENCE()
#endif

// round-to-nearest-even fp32 -> bf16 bits
static __device__ __forceinline__ unsigned short f2bf(float f) {
    unsigned u = __float_as_uint(f);
    u += 0x7FFFu + ((u >> 16) & 1u);
    return (unsigned short)(u >> 16);
}

static __device__ __forceinline__ v16bf make_frag(uint4 lo, uint4 hi) {
    union { uint4 q[2]; v16bf v; } u;
    u.q[0] = lo; u.q[1] = hi;
    return u.v;
}

// ---- prep: W1 [D][H] fp32 -> W1T [H][D] bf16 ; W2 [H][E] fp32 -> W2T [E][H] bf16
__global__ void prep_weights(const float* __restrict__ W1, const float* __restrict__ W2,
                             unsigned short* __restrict__ W1T, unsigned short* __restrict__ W2T) {
    const int n1 = H_MID * D_IN;
    const int n2 = E_EXP * H_MID;
    for (int i = blockIdx.x * blockDim.x + threadIdx.x; i < n1 + n2;
         i += gridDim.x * blockDim.x) {
        if (i < n1) {
            int h = i / D_IN, d = i - h * D_IN;
            W1T[i] = f2bf(W1[d * H_MID + h]);
        } else {
            int j = i - n1;
            int e = j / H_MID, h = j - e * H_MID;
            W2T[j] = f2bf(W2[h * E_EXP + e]);
        }
    }
}

// ---- fused gate: 32 rows per block, 8 wave32s; exactly 1 workgroup (2 waves/SIMD)
// is the intended residency, so give the scheduler a ~512-VGPR budget.
__global__ void __launch_bounds__(256, 2)
gate_fused(const float* __restrict__ x,
           const float* __restrict__ b1, const float* __restrict__ g1, const float* __restrict__ be1,
           const float* __restrict__ b2, const float* __restrict__ g2, const float* __restrict__ be2,
           const unsigned short* __restrict__ W1T, const unsigned short* __restrict__ W2T,
           float* __restrict__ out_w, int* __restrict__ out_i)
{
    // union buffer: phase1 = double-buffered x stage; phase2 = LN scratch;
    // then h bf16 [32][1024] (64KB); then logits f32 [32][64]
    __shared__ __align__(16) unsigned char smem[65536];

    const int tid  = threadIdx.x;
    const int lane = tid & 31;
    const int w    = tid >> 5;
    const int hh   = lane >> 4;    // half-wave (K-run selector)
    const int ln   = lane & 15;
    const int rowBase = blockIdx.x * 32;

    // ---------------- Phase 1: acc = x @ W1  (M=32, N=1024, K=2048, bf16 WMMA)
    const int r = w & 1;           // row tile (16 rows)
    const int c = w >> 1;          // column group: [c*256, c*256+256)

    v8f acc[16];
#pragma unroll
    for (int j = 0; j < 16; ++j) acc[j] = {};

    const int m_l = r * 16 + ln;

    // prologue: stage k0=0 chunk into buffer 0
    {
        int i = tid * 4, m = i >> 5, k = i & 31;
        const float* gp = x + (size_t)(rowBase + m) * D_IN + k;
#if USE_ASYNC_LDS
        __builtin_amdgcn_global_load_async_to_lds_b128(
            (AS1 v4i*)gp, (AS3 v4i*)(smem + m * 128 + k * 4), 0, 0);
#else
        v4f v = __builtin_nontemporal_load((const v4f*)gp);
        unsigned p0 = (unsigned)f2bf(v[0]) | ((unsigned)f2bf(v[1]) << 16);
        unsigned p1 = (unsigned)f2bf(v[2]) | ((unsigned)f2bf(v[3]) << 16);
        *(uint2*)(smem + m * 64 + k * 2) = make_uint2(p0, p1);
#endif
    }

    for (int k0 = 0; k0 < D_IN; k0 += 32) {
#if USE_ASYNC_LDS
        __builtin_amdgcn_s_wait_asynccnt(0);   // this wave's async stores landed in LDS
#endif
        __syncthreads();                       // current buffer visible to all waves
        const int cur = (k0 >> 5) & 1;

        if (k0 + 32 < D_IN) {                  // stage next chunk into other buffer
            int i = tid * 4, m = i >> 5, k = i & 31;
            const float* gp = x + (size_t)(rowBase + m) * D_IN + (k0 + 32) + k;
#if USE_ASYNC_LDS
            __builtin_amdgcn_global_load_async_to_lds_b128(
                (AS1 v4i*)gp,
                (AS3 v4i*)(smem + (cur ^ 1) * 4096 + m * 128 + k * 4), 0, 0);
#else
            v4f v = __builtin_nontemporal_load((const v4f*)gp);
            unsigned p0 = (unsigned)f2bf(v[0]) | ((unsigned)f2bf(v[1]) << 16);
            unsigned p1 = (unsigned)f2bf(v[2]) | ((unsigned)f2bf(v[3]) << 16);
            *(uint2*)(smem + (cur ^ 1) * 2048 + m * 64 + k * 2) = make_uint2(p0, p1);
#endif
        }

        // B fragments: preload ALL 16 tiles (32 b128 loads) into distinct registers;
        // loads return in order, so each WMMA then needs only a partial s_wait_loadcnt.
        const unsigned short* bbase = W1T + k0 + hh * 8 + (size_t)(c * 256 + ln) * D_IN;
        v16bf bq[16];
#pragma unroll
        for (int p = 0; p < 16; ++p) {
            const unsigned short* bp = bbase + (size_t)p * 16 * D_IN;
            bq[p] = make_frag(*(const uint4*)bp, *(const uint4*)(bp + 16));
        }

        // A fragment (16x32 bf16): lane half hh -> K runs {hh*8.., 16+hh*8..}
#if USE_ASYNC_LDS
        const unsigned char* abase = smem + cur * 4096 + m_l * 128;
        v4f f0 = *(const v4f*)(abase + hh * 32);
        v4f f1 = *(const v4f*)(abase + hh * 32 + 16);
        v4f f2 = *(const v4f*)(abase + 64 + hh * 32);
        v4f f3 = *(const v4f*)(abase + 64 + hh * 32 + 16);
        union { unsigned short s[16]; v16bf v; } ua;
#pragma unroll
        for (int p = 0; p < 4; ++p) {
            ua.s[p]      = f2bf(f0[p]);
            ua.s[4 + p]  = f2bf(f1[p]);
            ua.s[8 + p]  = f2bf(f2[p]);
            ua.s[12 + p] = f2bf(f3[p]);
        }
        v16bf a = ua.v;
#else
        uint4 alo = *(const uint4*)(smem + cur * 2048 + m_l * 64 + hh * 16);
        uint4 ahi = *(const uint4*)(smem + cur * 2048 + m_l * 64 + 32 + hh * 16);
        v16bf a = make_frag(alo, ahi);
#endif

        // keep all fragment loads above, all WMMAs below: forces in-flight pipeline
        SCHED_FENCE();

#pragma unroll
        for (int j = 0; j < 16; ++j) {
            acc[j] = __builtin_amdgcn_wmma_f32_16x16x32_bf16(
                         false, a, false, bq[j], (short)0, acc[j], false, false);
        }
    }
    __syncthreads();   // everyone done with staging buffers before reuse

    // ---------------- Phase 2: +b1, LayerNorm over H, ReLU, store bf16 h to LDS
    float s[8], q[8];
#pragma unroll
    for (int g = 0; g < 8; ++g) { s[g] = 0.f; q[g] = 0.f; }
#pragma unroll
    for (int j = 0; j < 16; ++j) {
        float bias = b1[c * 256 + j * 16 + ln];
#pragma unroll
        for (int g = 0; g < 8; ++g) {
            float v = acc[j][g] + bias;
            acc[j][g] = v;
            s[g] += v;
            q[g] += v * v;
        }
    }
#pragma unroll
    for (int off = 8; off >= 1; off >>= 1) {
#pragma unroll
        for (int g = 0; g < 8; ++g) {
            s[g] += __shfl_xor(s[g], off, 32);
            q[g] += __shfl_xor(q[g], off, 32);
        }
    }
    float* red = (float*)smem;               // [2][4][32] partials
    if (ln == 0) {
#pragma unroll
        for (int g = 0; g < 8; ++g) {
            int row = r * 16 + hh * 8 + g;
            red[c * 32 + row]       = s[g];
            red[128 + c * 32 + row] = q[g];
        }
    }
    __syncthreads();
    float* stat = (float*)(smem + 2048);     // mu[32], rsigma[32]
    if (tid < 32) {
        float S = 0.f, Q = 0.f;
#pragma unroll
        for (int cc = 0; cc < 4; ++cc) {
            S += red[cc * 32 + tid];
            Q += red[128 + cc * 32 + tid];
        }
        float mu  = S * (1.0f / H_MID);
        float var = Q * (1.0f / H_MID) - mu * mu;
        stat[tid]      = mu;
        stat[32 + tid] = rsqrtf(var + LN_EPSF);
    }
    __syncthreads();
    float mu_r[8], rs_r[8];
#pragma unroll
    for (int g = 0; g < 8; ++g) {
        int row = r * 16 + hh * 8 + g;
        mu_r[g] = stat[row];
        rs_r[g] = stat[32 + row];
    }
    __syncthreads();   // stats captured; smem becomes h [32][1024] bf16

#pragma unroll
    for (int j = 0; j < 16; ++j) {
        int col = c * 256 + j * 16 + ln;
        float gam = g1[col];
        float bet = be1[col];
#pragma unroll
        for (int g = 0; g < 8; ++g) {
            int row = r * 16 + hh * 8 + g;
            float v = (acc[j][g] - mu_r[g]) * rs_r[g] * gam + bet;
            v = fmaxf(v, 0.0f);                            // ReLU
            *(unsigned short*)(smem + row * 2048 + col * 2) = f2bf(v);
        }
    }
    __syncthreads();

    // ---------------- Phase 3: logits = h @ W2  (M=32, N=64, K=1024), depth-2 pipeline
    const int rw = w & 1;
    const int cw = w >> 1;
    v8f acc2 = {};
    const int m2 = rw * 16 + ln;
    const unsigned short* w2base = W2T + (size_t)(cw * 16 + ln) * H_MID + hh * 8;
    v16bf b2f[2];
#pragma unroll
    for (int p = 0; p < 2; ++p)
        b2f[p] = make_frag(*(const uint4*)(w2base + p * 32),
                           *(const uint4*)(w2base + p * 32 + 16));
#pragma unroll
    for (int k0 = 0; k0 < H_MID; k0 += 32) {
        uint4 alo = *(const uint4*)(smem + m2 * 2048 + k0 * 2 + hh * 16);
        uint4 ahi = *(const uint4*)(smem + m2 * 2048 + k0 * 2 + 32 + hh * 16);
        v16bf a = make_frag(alo, ahi);
        v16bf bc = b2f[(k0 >> 5) & 1];
        if (k0 + 64 < H_MID)
            b2f[(k0 >> 5) & 1] = make_frag(*(const uint4*)(w2base + k0 + 64),
                                           *(const uint4*)(w2base + k0 + 64 + 16));
        acc2 = __builtin_amdgcn_wmma_f32_16x16x32_bf16(
                   false, a, false, bc, (short)0, acc2, false, false);
    }
    __syncthreads();   // done reading h

    float* lg = (float*)smem;                // logits [32][64] f32
    {
        float bias2 = b2[cw * 16 + ln];
#pragma unroll
        for (int g = 0; g < 8; ++g) {
            int row = rw * 16 + hh * 8 + g;
            lg[row * 64 + cw * 16 + ln] = acc2[g] + bias2;
        }
    }
    __syncthreads();

    // ---------------- Phase 4: LN(E) + softmax + top-8 + renorm + scatter
    for (int rr = 0; rr < 4; ++rr) {
        int row = w * 4 + rr;
        int c0 = lane, c1 = lane + 32;
        float v0 = lg[row * 64 + c0];
        float v1 = lg[row * 64 + c1];

        float sm = v0 + v1;
#pragma unroll
        for (int off = 16; off >= 1; off >>= 1) sm += __shfl_xor(sm, off, 32);
        float mu = sm * (1.0f / E_EXP);
        float d0 = v0 - mu, d1 = v1 - mu;
        float qq = d0 * d0 + d1 * d1;
#pragma unroll
        for (int off = 16; off >= 1; off >>= 1) qq += __shfl_xor(qq, off, 32);
        float rs = rsqrtf(qq * (1.0f / E_EXP) + LN_EPSF);
        float z0 = d0 * rs * g2[c0] + be2[c0];
        float z1 = d1 * rs * g2[c1] + be2[c1];

        float mx = fmaxf(z0, z1);
#pragma unroll
        for (int off = 16; off >= 1; off >>= 1) mx = fmaxf(mx, __shfl_xor(mx, off, 32));
        float e0 = __expf(z0 - mx), e1 = __expf(z1 - mx);
        float Z = e0 + e1;
#pragma unroll
        for (int off = 16; off >= 1; off >>= 1) Z += __shfl_xor(Z, off, 32);
        float p0 = e0 / Z, p1 = e1 / Z;

        float tv[K_TOP]; int ti[K_TOP];
        float a0 = p0, a1 = p1;
#pragma unroll
        for (int t = 0; t < K_TOP; ++t) {
            float v = a0; int id = c0;
            if (a1 > v) { v = a1; id = c1; }
#pragma unroll
            for (int off = 16; off >= 1; off >>= 1) {
                float ov = __shfl_xor(v, off, 32);
                int   oi = __shfl_xor(id, off, 32);
                if (ov > v || (ov == v && oi < id)) { v = ov; id = oi; }
            }
            tv[t] = v; ti[t] = id;
            if (id == c0) a0 = -1.0f;
            if (id == c1) a1 = -1.0f;
        }
        float ts = 0.f;
#pragma unroll
        for (int t = 0; t < K_TOP; ++t) ts += tv[t];
        float rn = 1.0f / (ts + 1e-8f);

        float w0 = 0.f, w1 = 0.f;
#pragma unroll
        for (int t = 0; t < K_TOP; ++t) {
            if (c0 == ti[t]) w0 = tv[t] * rn;
            if (c1 == ti[t]) w1 = tv[t] * rn;
        }
        size_t grow = (size_t)(rowBase + row);
        __builtin_nontemporal_store(w0, &out_w[grow * E_EXP + c0]);
        __builtin_nontemporal_store(w1, &out_w[grow * E_EXP + c1]);
#pragma unroll
        for (int t = 0; t < K_TOP; ++t)
            if (lane == t) __builtin_nontemporal_store(ti[t], &out_i[grow * K_TOP + t]);
    }
}

extern "C" void kernel_launch(void* const* d_in, const int* in_sizes, int n_in,
                              void* d_out, int out_size, void* d_ws, size_t ws_size,
                              hipStream_t stream) {
    (void)in_sizes; (void)n_in; (void)out_size; (void)ws_size;
    const float* x   = (const float*)d_in[0];
    const float* W1  = (const float*)d_in[1];
    const float* b1  = (const float*)d_in[2];
    const float* g1  = (const float*)d_in[3];
    const float* be1 = (const float*)d_in[4];
    const float* W2  = (const float*)d_in[5];
    const float* b2  = (const float*)d_in[6];
    const float* g2  = (const float*)d_in[7];
    const float* be2 = (const float*)d_in[8];

    unsigned short* W1T = (unsigned short*)d_ws;                 // [H][D] bf16, 4 MB
    unsigned short* W2T = W1T + (size_t)H_MID * D_IN;            // [E][H] bf16, 128 KB

    prep_weights<<<2048, 256, 0, stream>>>(W1, W2, W1T, W2T);

    float* out_w = (float*)d_out;                                // [B][E] fp32
    int*   out_i = (int*)(out_w + (size_t)B_ROWS * E_EXP);       // [B][8] int32
    gate_fused<<<B_ROWS / 32, 256, 0, stream>>>(x, b1, g1, be1, b2, g2, be2,
                                                W1T, W2T, out_w, out_i);
}
